// attention_61443802136876
// MI455X (gfx1250) — matrix operand order repaired and checked by reading, not compiled, not run on hardware
//
#include <hip/hip_runtime.h>

typedef __attribute__((ext_vector_type(16))) __bf16 v16bf;
typedef __attribute__((ext_vector_type(8)))  __bf16 v8bf;
typedef __attribute__((ext_vector_type(4)))  __bf16 v4bf;
typedef __attribute__((ext_vector_type(8)))  float  v8f;
typedef __attribute__((ext_vector_type(4)))  unsigned int v4u;
typedef __attribute__((ext_vector_type(8)))  int v8i;
typedef __attribute__((ext_vector_type(4)))  int v4i;

#define N_ROWS 8192
#define D_IN   1024
#define D_OUT  1024

// ---- WMMA fragment loaders (CDNA5 16-bit operand layouts, wave32) ----
// A (16x32, MxK): lane m = lane&15, half = lane>>4.
//   VGPR 0..3 : K = half*8 + {0,1},{2,3},{4,5},{6,7}
//   VGPR 4..7 : K = 16 + half*8 + {0..7}
__device__ __forceinline__ v16bf frag_a(const __bf16* p, int half) {
  v16bf f;
  *(v8bf*)&f       = *(const v8bf*)(p + half * 8);
  *((v8bf*)&f + 1) = *(const v8bf*)(p + half * 8 + 16);
  return f;
}
// B (32x16, KxN) from a [N][K] (transposed) buffer: lane n = lane&15.
//   same K order as A: VGPR 0..3 K = half*8 + {0..7}, VGPR 4..7 K = 16 + half*8 + {0..7}
__device__ __forceinline__ v16bf frag_b(const __bf16* p, int half) {
  v16bf f;
  *(v8bf*)&f       = *(const v8bf*)(p + half * 8);
  *((v8bf*)&f + 1) = *(const v8bf*)(p + half * 8 + 16);
  return f;
}
__device__ __forceinline__ v8f wmma_bf16(v16bf a, v16bf b, v8f c) {
  return __builtin_amdgcn_wmma_f32_16x16x32_bf16(false, a, false, b, (short)0, c,
                                                 false, false);
}

// ---- Tensor Data Mover: 2D bf16 tile (tile_d0 x tile_d1) global -> LDS ----
// D# per CDNA5 ISA 8.3/8.4: group0 {count=1, lds_addr, global_addr, type=2},
// group1 {data_size=2B, tensor dims (never clip), tile dims, dim0 stride}.
// This toolchain exposes the 6-arg builtin (groups 2/3 + extra group, cpol).
__device__ __forceinline__ unsigned lds_off(const void* p) {
  return (unsigned)(unsigned long long)p;   // LDS aperture: addr[31:0] = LDS offset
}
__device__ __forceinline__ void tdm_load_2d_bf16(unsigned lds_addr, const void* gptr,
                                                 unsigned tensor_d0, unsigned tensor_d1,
                                                 unsigned tile_d0, unsigned tile_d1,
                                                 unsigned stride_elems) {
  unsigned long long ga = (unsigned long long)gptr;
  v4u g0;
  g0[0] = 1u;                                             // count=1 (valid, user mode)
  g0[1] = lds_addr;                                       // [63:32] lds_addr
  g0[2] = (unsigned)(ga & 0xffffffffu);                   // [95:64] global_addr lo
  g0[3] = (unsigned)((ga >> 32) & 0x01ffffffu) | (2u << 30);  // addr[56:32] | type=2
  v8i g1;
  g1[0] = (int)(1u << 16);                                // data_size=1 -> 2 bytes
  g1[1] = (int)((tensor_d0 & 0xffffu) << 16);             // tensor_dim0[15:0]
  g1[2] = (int)((tensor_d0 >> 16) | ((tensor_d1 & 0xffffu) << 16));
  g1[3] = (int)((tensor_d1 >> 16) | (tile_d0 << 16));     // tile_dim0
  g1[4] = (int)(tile_d1 & 0xffffu);                       // tile_dim1 (tile_dim2=0)
  g1[5] = (int)stride_elems;                              // tensor_dim0_stride lo32
  g1[6] = 0;
  g1[7] = 0;
  v4i z4 = {0, 0, 0, 0};                                  // groups 2/3: 2D tensor
  v8i z8 = {0, 0, 0, 0, 0, 0, 0, 0};
  __builtin_amdgcn_tensor_load_to_lds(g0, g1, z4, z4, z8, 0);
}

// ---- Pass 0a: feat f32 -> bf16 ----
__global__ __launch_bounds__(256) void cvt_feat_kernel(const float* __restrict__ f,
                                                       __bf16* __restrict__ o) {
  int i = (blockIdx.x * 256 + threadIdx.x) * 4;
  float4 v = *(const float4*)(f + i);
  v4bf t = { (__bf16)v.x, (__bf16)v.y, (__bf16)v.z, (__bf16)v.w };
  *(v4bf*)(o + i) = t;
}

// ---- Pass 0b: W f32 [K][N] -> bf16 transposed [N][K] (WMMA-B friendly) ----
__global__ __launch_bounds__(256) void cvt_w_kernel(const float* __restrict__ Wq,
                                                    const float* __restrict__ Wk,
                                                    const float* __restrict__ Wv,
                                                    __bf16* __restrict__ wt) {
  int z = blockIdx.z;
  const float* W = (z == 0) ? Wq : (z == 1) ? Wk : Wv;
  int n = blockIdx.x * 16 + threadIdx.x;
  int k = blockIdx.y * 16 + threadIdx.y;
  wt[(size_t)z * D_IN * D_OUT + (size_t)n * D_IN + k] = (__bf16)W[(size_t)k * D_OUT + n];
}

// ---- Pass 1: fused QKV projection. block = 128x64 tile, 8 waves.
//      TDM double-buffered staging: wave 0 DMAs k-step i+1 while all waves
//      run WMMAs on k-step i. Q pre-scaled by 1/sqrt(d); V stored vT[d][key]. ----
__global__ __launch_bounds__(256) void qkv_proj_kernel(
    const __bf16* __restrict__ feat, const __bf16* __restrict__ wt,
    const float* __restrict__ bq, const float* __restrict__ bk,
    const float* __restrict__ bv,
    __bf16* __restrict__ qb, __bf16* __restrict__ kb, __bf16* __restrict__ vtb) {
  const int tid  = threadIdx.x;
  const int wave = tid >> 5;
  const int lane = tid & 31;
  const int half = lane >> 4;
  const int ln16 = lane & 15;
  const int z    = blockIdx.z;
  const int rowbase = blockIdx.x * 128;
  const int colbase = blockIdx.y * 64;
  const __bf16* w   = wt + (size_t)z * D_IN * D_OUT;
  const float* bias = (z == 0) ? bq : (z == 1) ? bk : bv;

  __shared__ __bf16 As[2][128 * 32];   // 2 x 8 KB, [row][k] per buffer
  __shared__ __bf16 Bs[2][64 * 32];    // 2 x 4 KB, [n][k] per buffer

  v8f acc[4] = {};

  // prologue: DMA k-step 0 into buffer 0
  if (wave == 0) {
    tdm_load_2d_bf16(lds_off(&As[0][0]), feat + (size_t)rowbase * D_IN,
                     D_IN, N_ROWS, 32, 128, D_IN);
    tdm_load_2d_bf16(lds_off(&Bs[0][0]), w + (size_t)colbase * D_IN,
                     D_IN, D_OUT, 32, 64, D_IN);
    __builtin_amdgcn_s_wait_tensorcnt(0);
  }
  __syncthreads();

  for (int kk = 0; kk < D_IN; kk += 32) {
    const int cur = (kk >> 5) & 1;
    if (wave == 0 && kk + 32 < D_IN) {   // DMA next k-step into the other buffer
      tdm_load_2d_bf16(lds_off(&As[cur ^ 1][0]),
                       feat + (size_t)rowbase * D_IN + kk + 32,
                       D_IN, N_ROWS, 32, 128, D_IN);
      tdm_load_2d_bf16(lds_off(&Bs[cur ^ 1][0]),
                       w + (size_t)colbase * D_IN + kk + 32,
                       D_IN, D_OUT, 32, 64, D_IN);
    }
    v16bf a = frag_a(&As[cur][(wave * 16 + ln16) * 32], half);
#pragma unroll
    for (int nt = 0; nt < 4; ++nt) {
      v16bf b = frag_b(&Bs[cur][(nt * 16 + ln16) * 32], half);
      acc[nt] = wmma_bf16(a, b, acc[nt]);
    }
    if (wave == 0) __builtin_amdgcn_s_wait_tensorcnt(0);
    __syncthreads();
  }

  const float qscale = 0.03125f;   // 1/sqrt(1024), folded into Q
#pragma unroll
  for (int nt = 0; nt < 4; ++nt) {
    const int col = colbase + nt * 16 + ln16;
    const float badd = bias[col];
#pragma unroll
    for (int r = 0; r < 8; ++r) {
      const int row = rowbase + wave * 16 + half * 8 + r;
      float val = acc[nt][r] + badd;
      if (z == 0)      qb[(size_t)row * D_OUT + col]  = (__bf16)(val * qscale);
      else if (z == 1) kb[(size_t)row * D_OUT + col]  = (__bf16)val;
      else             vtb[(size_t)col * N_ROWS + row] = (__bf16)val;  // vT[d][key]
    }
  }
}

// ---- Pass 2: flash attention. Br=32 queries/block, Bc=64 keys/chunk.
//      K,V (32 MB bf16 total) stay L2-resident across the 256 blocks. ----
__global__ __launch_bounds__(256) void flash_attn_kernel(
    const __bf16* __restrict__ qb, const __bf16* __restrict__ kb,
    const __bf16* __restrict__ vtb, float* __restrict__ out) {
  const int tid  = threadIdx.x;
  const int wave = tid >> 5;
  const int lane = tid & 31;
  const int half = lane >> 4;
  const int ln16 = lane & 15;
  const int qbase = blockIdx.x * 32;

  const int mts = wave >> 2;   // S-phase: wave -> one 16x16 tile of the 32x64 S
  const int nts = wave & 3;

  __shared__ float  Ss[32 * 64];   // 8 KB raw scores
  __shared__ __bf16 Ps[32 * 64];   // 4 KB exp(S - m)
  __shared__ float  scaleS[32];
  __shared__ float  invlS[32];

  v8f accO[2][8] = {};             // wave owns 32 rows x 128 d-cols of O
  float m_i = -1e30f, l_i = 0.f;

  for (int keybase = 0; keybase < N_ROWS; keybase += 64) {
    // --- S tile: q @ k^T (A from Q, B from K rows: both straight b128s) ---
    v8f accS = {};
    const __bf16* qrow = qb + (size_t)(qbase + mts * 16 + ln16) * D_IN;
    const __bf16* krow = kb + (size_t)(keybase + nts * 16 + ln16) * D_IN;
    for (int kk = 0; kk < D_IN; kk += 32) {
      v16bf a = frag_a(qrow + kk, half);
      v16bf b = frag_b(krow + kk, half);
      accS = wmma_bf16(a, b, accS);
    }
#pragma unroll
    for (int r = 0; r < 8; ++r)
      Ss[(mts * 16 + half * 8 + r) * 64 + nts * 16 + ln16] = accS[r];
    __syncthreads();

    // --- online softmax: one row per thread (rows 0..31) ---
    if (tid < 32) {
      float mloc = m_i;
#pragma unroll 8
      for (int j = 0; j < 64; ++j) mloc = fmaxf(mloc, Ss[tid * 64 + j]);
      const float sc = __expf(m_i - mloc);
      float sum = 0.f;
#pragma unroll 8
      for (int j = 0; j < 64; ++j) {
        float p = __expf(Ss[tid * 64 + j] - mloc);
        sum += p;
        Ps[tid * 64 + j] = (__bf16)p;
      }
      l_i = l_i * sc + sum;
      m_i = mloc;
      scaleS[tid] = sc;
    }
    __syncthreads();

    // --- rescale running O, then O += P @ V (B from vT: contiguous K) ---
#pragma unroll
    for (int mt = 0; mt < 2; ++mt) {
      float sr[8];
#pragma unroll
      for (int r = 0; r < 8; ++r) sr[r] = scaleS[mt * 16 + half * 8 + r];
#pragma unroll
      for (int nt = 0; nt < 8; ++nt)
#pragma unroll
        for (int r = 0; r < 8; ++r) accO[mt][nt][r] *= sr[r];
    }
#pragma unroll
    for (int kc = 0; kc < 64; kc += 32) {
#pragma unroll
      for (int mt = 0; mt < 2; ++mt) {
        v16bf a = frag_a(&Ps[(mt * 16 + ln16) * 64 + kc], half);
#pragma unroll
        for (int nt = 0; nt < 8; ++nt) {
          const int dcol = wave * 128 + nt * 16 + ln16;
          v16bf b = frag_b(vtb + (size_t)dcol * N_ROWS + keybase + kc, half);
          accO[mt][nt] = wmma_bf16(a, b, accO[mt][nt]);
        }
      }
    }
  }

  // --- normalize by 1/l and write f32 output ---
  if (tid < 32) invlS[tid] = 1.0f / l_i;
  __syncthreads();
#pragma unroll
  for (int mt = 0; mt < 2; ++mt) {
    float inv[8];
#pragma unroll
    for (int r = 0; r < 8; ++r) inv[r] = invlS[mt * 16 + half * 8 + r];
#pragma unroll
    for (int nt = 0; nt < 8; ++nt) {
      const int col = wave * 128 + nt * 16 + ln16;
#pragma unroll
      for (int r = 0; r < 8; ++r) {
        const int row = qbase + mt * 16 + half * 8 + r;
        out[(size_t)row * D_OUT + col] = accO[mt][nt][r] * inv[r];
      }
    }
  }
}

extern "C" void kernel_launch(void* const* d_in, const int* in_sizes, int n_in,
                              void* d_out, int out_size, void* d_ws, size_t ws_size,
                              hipStream_t stream) {
  const float* feat = (const float*)d_in[0];
  const float* Wq   = (const float*)d_in[1];
  const float* bq   = (const float*)d_in[2];
  const float* Wk   = (const float*)d_in[3];
  const float* bk   = (const float*)d_in[4];
  const float* Wv   = (const float*)d_in[5];
  const float* bv   = (const float*)d_in[6];
  float* out = (float*)d_out;

  // workspace layout (70 MB total, 16B-aligned regions)
  char* ws = (char*)d_ws;
  __bf16* featb = (__bf16*)(ws);                          // 16 MB  feat bf16
  __bf16* wtb   = (__bf16*)(ws + (16u << 20));            //  6 MB  Wq/Wk/Wv^T bf16
  __bf16* qb    = (__bf16*)(ws + (22u << 20));            // 16 MB  q bf16 (pre-scaled)
  __bf16* kb    = (__bf16*)(ws + (38u << 20));            // 16 MB  k bf16
  __bf16* vtb   = (__bf16*)(ws + (54u << 20));            // 16 MB  v^T bf16

  cvt_feat_kernel<<<(N_ROWS * D_IN) / (256 * 4), 256, 0, stream>>>(feat, featb);
  cvt_w_kernel<<<dim3(D_OUT / 16, D_IN / 16, 3), dim3(16, 16), 0, stream>>>(Wq, Wk, Wv, wtb);
  qkv_proj_kernel<<<dim3(N_ROWS / 128, D_OUT / 64, 3), 256, 0, stream>>>(
      featb, wtb, bq, bk, bv, qb, kb, vtb);
  flash_attn_kernel<<<dim3(N_ROWS / 32), 256, 0, stream>>>(qb, kb, vtb, out);

  (void)in_sizes; (void)n_in; (void)out_size; (void)ws_size;
}
